// GraphEdgeConvEmb_11020886081779
// MI455X (gfx1250) — matrix operand order repaired and checked by reading, not compiled, not run on hardware
//
#include <hip/hip_runtime.h>
#include <math.h>

#define NN 50000
#define NE 600000
#define CC 128
#define NLAYER 5
#define FFD 512

typedef __attribute__((ext_vector_type(16))) __bf16 v16bf;
typedef __attribute__((ext_vector_type(8)))  float  v8f;
typedef __attribute__((ext_vector_type(16))) unsigned short v16u;
typedef __attribute__((ext_vector_type(8)))  unsigned short v8u;
typedef __attribute__((ext_vector_type(4)))  unsigned int   u32x4;
typedef __attribute__((ext_vector_type(8)))  int            i32x8;
typedef __attribute__((ext_vector_type(4)))  int            i32x4;

__device__ __forceinline__ unsigned short f2bf(float f) {
  return __builtin_bit_cast(unsigned short, (__bf16)f);   // native v_cvt, RNE
}
__device__ __forceinline__ float bf2f(unsigned short s) {
  return __uint_as_float(((unsigned int)s) << 16);
}

__device__ __forceinline__ v16bf pack16(v8u lo, v8u hi) {
  v16u r;
#pragma unroll
  for (int i = 0; i < 8; ++i) { r[i] = lo[i]; r[i + 8] = hi[i]; }
  return __builtin_bit_cast(v16bf, r);
}

// A fragment (16x32 bf16) per ISA 7.12.2: lanes 0-15 row M=lane hold K = k0+{0..7,16..23},
// lanes 16-31 row M=lane-16 hold K = k0+{8..15,24..31}.  Two 16B contiguous runs per lane.
__device__ __forceinline__ v16bf frag_a_bf16(const unsigned short* A, long row0, int ld,
                                             int k0, int lane) {
  int m = lane & 15;
  int koff = k0 + ((lane >> 4) << 3);
  const unsigned short* p = A + (row0 + m) * (long)ld + koff;
  v8u lo = *(const v8u*)p;
  v8u hi = *(const v8u*)(p + 16);
  return pack16(lo, hi);
}

__device__ __forceinline__ v16bf frag_a_f32(const float* A, long row0, int ld,
                                            int k0, int lane) {
  int m = lane & 15;
  int koff = k0 + ((lane >> 4) << 3);
  const float* p = A + (row0 + m) * (long)ld + koff;
  v16bf r;
#pragma unroll
  for (int i = 0; i < 8; ++i) { r[i] = (__bf16)p[i]; r[i + 8] = (__bf16)p[16 + i]; }
  return r;
}

// B fragment (32x16 bf16) from LDS holding B^T (Bt[n][k], 128 K per row):
// lane holds column n0+(lane&15); lanes<16 K=k0+{0..7,16..23}, lanes>=16 K=k0+{8..15,24..31}.
__device__ __forceinline__ v16bf frag_b(const unsigned short* Bt, int n0, int k0, int lane) {
  int n = n0 + (lane & 15);
  int kk = k0 + ((lane >> 4) << 3);
  const unsigned short* p = Bt + n * 128 + kk;
  v8u lo = *(const v8u*)p;
  v8u hi = *(const v8u*)(p + 16);
  return pack16(lo, hi);
}

__device__ __forceinline__ v8f wmma_bf(v16bf a, v16bf b, v8f c) {
  return __builtin_amdgcn_wmma_f32_16x16x32_bf16(false, a, false, b, (short)0, c,
                                                 false, false);
}

// ---- Tensor Data Mover: DMA one contiguous 32KB bf16 panel (128x128, pre-transposed
// in global) into LDS offset 0.  D# per CDNA5 ISA:
//   group0: count=1 | lds_addr=0 | global_addr | type=2
//   group1: data_size=1(2B), tensor = 1 row x 16384 elems, tile = 16384x1, stride 16384
#if __has_builtin(__builtin_amdgcn_tensor_load_to_lds)
#define USE_TDM 1
__device__ __forceinline__ void tdm_load_32k(const unsigned short* gsrc) {
  unsigned long long ga = (unsigned long long)gsrc;
  u32x4 g0;
  g0[0] = 1u;                                        // count=1 (valid user D#)
  g0[1] = 0u;                                        // lds_addr = 0 (Bt is first LDS symbol)
  g0[2] = (unsigned)(ga & 0xFFFFFFFFu);              // global_addr[31:0]
  g0[3] = (unsigned)((ga >> 32) & 0x01FFFFFFu) | (2u << 30);  // addr[56:32] | type=2
  i32x8 g1;
  g1[0] = (int)(1u << 16);                           // data_size=1 -> 2 bytes
  g1[1] = (int)(16384u << 16);                       // tensor_dim0[15:0] in [31:16]
  g1[2] = (int)(1u << 16);                           // tensor_dim0 msbs=0 | tensor_dim1=1
  g1[3] = (int)(16384u << 16);                       // tensor_dim1 msbs=0 | tile_dim0=16384
  g1[4] = 1;                                         // tile_dim1=1, tile_dim2=0
  g1[5] = 16384;                                     // tensor_dim0_stride[31:0]
  g1[6] = 0;
  g1[7] = 0;
  i32x4 z = {};
#if __clang_major__ >= 23
  i32x8 z8 = {};
  __builtin_amdgcn_tensor_load_to_lds(g0, g1, z, z, z8, 0);
#else
  __builtin_amdgcn_tensor_load_to_lds(g0, g1, z, z, 0);
#endif
}
#endif

// Stage pre-transposed weight panel (contiguous 16384 bf16) into LDS Bt.
// Caller must __syncthreads() afterwards.
__device__ __forceinline__ void stage_weight(const unsigned short* __restrict__ Wt,
                                             unsigned short* Bt, int tid) {
#ifdef USE_TDM
  if (tid < 32) {                    // one wave issues the DMA; TDM signals once
    tdm_load_32k(Wt);
    __builtin_amdgcn_s_wait_tensorcnt(0);
  }
  // The TDM writes LDS behind the compiler's alias analysis, and a bare asm
  // "memory" clobber cannot write a non-escaping LDS object (escape analysis
  // still folds loads of Bt to undef -> the whole GEMM collapses, observed).
  // So plant a *visible* store with an opaque, dynamically-false guard and an
  // unknown index: every later load of Bt must then really be performed.
  int never = 0;
  asm volatile("" : "+s"(never));    // opaque: compiler can't prove it stays 0
  if (never) Bt[never & (128 * 128 - 1)] = (unsigned short)never;
#else
  for (int t = tid; t < 128 * 128; t += 256) Bt[t] = Wt[t];
#endif
}

// ---------------- small utility kernels ----------------

__global__ void k_zero(float* p, int n) {
  int i = blockIdx.x * 256 + threadIdx.x;
  if (i < n) p[i] = 0.f;
}

__global__ void k_f2bf(const float* __restrict__ s, unsigned short* __restrict__ d, int n) {
  int i = blockIdx.x * 256 + threadIdx.x;
  if (i < n) d[i] = f2bf(s[i]);
}

// Transpose-convert weight W[k][n] (f32, 128 x ncols) -> Wt[n][k] (bf16, ncols x 128)
__global__ void k_f2bf_t(const float* __restrict__ s, unsigned short* __restrict__ d,
                         int ncols) {
  int t = blockIdx.x * 256 + threadIdx.x;
  if (t >= ncols * 128) return;
  int n = t >> 7, k = t & 127;
  d[t] = f2bf(s[(long)k * ncols + n]);
}

// h = x @ W_vert   [N,16]@[16,128] -> f32 + bf16
__global__ void k_vert(const float* __restrict__ x, const float* __restrict__ Wv,
                       float* __restrict__ h, unsigned short* __restrict__ hbf) {
  __shared__ float xs[16];
  int n = blockIdx.x, c = threadIdx.x;
  if (c < 16) xs[c] = x[(long)n * 16 + c];
  __syncthreads();
  float s = 0.f;
#pragma unroll
  for (int d = 0; d < 16; ++d) s += xs[d] * Wv[d * CC + c];
  long o = (long)n * CC + c;
  h[o] = s;
  hbf[o] = f2bf(s);
}

// ea_bf = bf16(edge_attr @ W_edge_ff)   [E,128]@[128,128]
__global__ void k_edge_ff(const float* __restrict__ A, const unsigned short* __restrict__ Wt,
                          unsigned short* __restrict__ eabf, int M) {
  __shared__ unsigned short Bt[128 * 128];
  stage_weight(Wt, Bt, threadIdx.x);
  __syncthreads();
  int wave = threadIdx.x >> 5, lane = threadIdx.x & 31;
  long row = ((long)blockIdx.x * 8 + wave) * 16;
  if (row >= M) return;
  v16bf a[4];
#pragma unroll
  for (int k = 0; k < 4; ++k) a[k] = frag_a_f32(A, row, CC, k * 32, lane);
  int col0 = lane & 15, mo = (lane >> 4) << 3;
#pragma unroll
  for (int n = 0; n < 8; ++n) {
    v8f acc = {};
#pragma unroll
    for (int k = 0; k < 4; ++k) acc = wmma_bf(a[k], frag_b(Bt, n * 16, k * 32, lane), acc);
    int col = n * 16 + col0;
#pragma unroll
    for (int r = 0; r < 8; ++r) eabf[(row + mo + r) * (long)CC + col] = f2bf(acc[r]);
  }
}

// eaWe = ea_bf @ We ; msg = h[src]*eaWe ; atomic scatter-add into agg[dst]
__global__ void k_edge_scatter(const unsigned short* __restrict__ eabf,
                               const unsigned short* __restrict__ Wet,
                               const float* __restrict__ h,
                               const int* __restrict__ src, const int* __restrict__ dst,
                               float* __restrict__ agg, int M) {
  __shared__ unsigned short Bt[128 * 128];
  stage_weight(Wet, Bt, threadIdx.x);
  __syncthreads();
  int wave = threadIdx.x >> 5, lane = threadIdx.x & 31;
  long row = ((long)blockIdx.x * 8 + wave) * 16;
  if (row >= M) return;
  // warm L2 for the next tile of the edge stream (dominant HBM traffic)
  __builtin_prefetch(eabf + (row + 128) * (long)CC, 0, 1);
  v16bf a[4];
#pragma unroll
  for (int k = 0; k < 4; ++k) a[k] = frag_a_bf16(eabf, row, CC, k * 32, lane);
  int col0 = lane & 15, mo = (lane >> 4) << 3;
#pragma unroll
  for (int n = 0; n < 8; ++n) {
    v8f acc = {};
#pragma unroll
    for (int k = 0; k < 4; ++k) acc = wmma_bf(a[k], frag_b(Bt, n * 16, k * 32, lane), acc);
    int col = n * 16 + col0;
#pragma unroll
    for (int r = 0; r < 8; ++r) {
      long e = row + mo + r;
      int s = src[e], d = dst[e];
      float msg = acc[r] * h[(long)s * CC + col];
      atomicAdd(&agg[(long)d * CC + col], msg);
    }
  }
}

// out = relu(agg_bf@Wr + h_bf@Wro + br + bro), accumulate per-channel sum/sumsq
__global__ void k_node(const unsigned short* __restrict__ aggbf,
                       const unsigned short* __restrict__ Wrt,
                       const unsigned short* __restrict__ hbf,
                       const unsigned short* __restrict__ Wrot,
                       const float* __restrict__ br, const float* __restrict__ bro,
                       float* __restrict__ outb, float* __restrict__ bnsum,
                       float* __restrict__ bnsq, int M) {
  __shared__ unsigned short Bt[128 * 128];   // first LDS symbol -> offset 0 for TDM
  __shared__ float s1[CC], s2[CC];
  int tid = threadIdx.x;
  if (tid < CC) { s1[tid] = 0.f; s2[tid] = 0.f; }
  stage_weight(Wrt, Bt, tid);
  __syncthreads();
  int wave = tid >> 5, lane = tid & 31;
  long row = ((long)blockIdx.x * 8 + wave) * 16;
  bool active = row < M;
  v8f acc[8];
  v16bf a[4];
  if (active) {
#pragma unroll
    for (int k = 0; k < 4; ++k) a[k] = frag_a_bf16(aggbf, row, CC, k * 32, lane);
#pragma unroll
    for (int n = 0; n < 8; ++n) {
      v8f t = {};
#pragma unroll
      for (int k = 0; k < 4; ++k) t = wmma_bf(a[k], frag_b(Bt, n * 16, k * 32, lane), t);
      acc[n] = t;
    }
  }
  __syncthreads();
  stage_weight(Wrot, Bt, tid);
  __syncthreads();
  if (active) {
#pragma unroll
    for (int k = 0; k < 4; ++k) a[k] = frag_a_bf16(hbf, row, CC, k * 32, lane);
    int col0 = lane & 15, mo = (lane >> 4) << 3;
#pragma unroll
    for (int n = 0; n < 8; ++n) {
      v8f t = acc[n];
#pragma unroll
      for (int k = 0; k < 4; ++k) t = wmma_bf(a[k], frag_b(Bt, n * 16, k * 32, lane), t);
      int col = n * 16 + col0;
      float bias = br[col] + bro[col];
      float ls = 0.f, lq = 0.f;
#pragma unroll
      for (int r = 0; r < 8; ++r) {
        float v = t[r] + bias;
        v = v > 0.f ? v : 0.f;
        outb[(row + mo + r) * (long)CC + col] = v;
        ls += v;
        lq += v * v;
      }
      atomicAdd(&s1[col], ls);
      atomicAdd(&s2[col], lq);
    }
  }
  __syncthreads();
  if (tid < CC) {
    atomicAdd(&bnsum[tid], s1[tid]);
    atomicAdd(&bnsq[tid], s2[tid]);
  }
}

__global__ void k_bn_final(const float* __restrict__ bnsum, const float* __restrict__ bnsq,
                           const float* __restrict__ gamma, const float* __restrict__ beta,
                           float* __restrict__ scale, float* __restrict__ shift, float invN) {
  int c = threadIdx.x;
  float mu = bnsum[c] * invN;
  float var = bnsq[c] * invN - mu * mu;
  float sc = gamma[c] * rsqrtf(var + 1e-5f);
  scale[c] = sc;
  shift[c] = beta[c] - mu * sc;
}

// h = BN(out)*scale + shift + h_bf@Wrs ; rewrite h (f32 + bf16) in place (row-private)
__global__ void k_resid(const unsigned short* __restrict__ hbf_in,
                        const unsigned short* __restrict__ Wrst,
                        const float* __restrict__ outb,
                        const float* __restrict__ scale, const float* __restrict__ shift,
                        float* __restrict__ h_new, unsigned short* __restrict__ hbf_new,
                        int M) {
  __shared__ unsigned short Bt[128 * 128];
  stage_weight(Wrst, Bt, threadIdx.x);
  __syncthreads();
  int wave = threadIdx.x >> 5, lane = threadIdx.x & 31;
  long row = ((long)blockIdx.x * 8 + wave) * 16;
  if (row >= M) return;
  v16bf a[4];
#pragma unroll
  for (int k = 0; k < 4; ++k) a[k] = frag_a_bf16(hbf_in, row, CC, k * 32, lane);
  int col0 = lane & 15, mo = (lane >> 4) << 3;
#pragma unroll
  for (int n = 0; n < 8; ++n) {
    v8f acc = {};
#pragma unroll
    for (int k = 0; k < 4; ++k) acc = wmma_bf(a[k], frag_b(Bt, n * 16, k * 32, lane), acc);
    int col = n * 16 + col0;
    float sc = scale[col], sh = shift[col];
#pragma unroll
    for (int r = 0; r < 8; ++r) {
      long o = (row + mo + r) * (long)CC + col;
      float v = outb[o] * sc + sh + acc[r];
      h_new[o] = v;
      hbf_new[o] = f2bf(v);
    }
  }
}

// hh = gelu(h_bf @ W_h1 + b1)  -> bf16 [N,512]; column chunk via blockIdx.y
__global__ void k_head(const unsigned short* __restrict__ hbf,
                       const unsigned short* __restrict__ W1t,   // [512][128] bf16
                       const float* __restrict__ b1,
                       unsigned short* __restrict__ hhbf, int M) {
  __shared__ unsigned short Bt[128 * 128];
  int c0 = blockIdx.y * 128;
  stage_weight(W1t + (long)c0 * 128, Bt, threadIdx.x);   // contiguous chunk
  __syncthreads();
  int wave = threadIdx.x >> 5, lane = threadIdx.x & 31;
  long row = ((long)blockIdx.x * 8 + wave) * 16;
  if (row >= M) return;
  v16bf a[4];
#pragma unroll
  for (int k = 0; k < 4; ++k) a[k] = frag_a_bf16(hbf, row, CC, k * 32, lane);
  int col0 = lane & 15, mo = (lane >> 4) << 3;
#pragma unroll
  for (int n = 0; n < 8; ++n) {
    v8f acc = {};
#pragma unroll
    for (int k = 0; k < 4; ++k) acc = wmma_bf(a[k], frag_b(Bt, n * 16, k * 32, lane), acc);
    int col = c0 + n * 16 + col0;
    float bias = b1[col];
#pragma unroll
    for (int r = 0; r < 8; ++r) {
      float v = acc[r] + bias;
      float g = 0.5f * v * (1.f + erff(v * 0.70710678118654752f));  // exact GELU
      hhbf[(row + mo + r) * (long)FFD + col] = f2bf(g);
    }
  }
}

// out[n] = hh[n,:] . W_h2 + b2
__global__ void k_head2(const unsigned short* __restrict__ hhbf,
                        const float* __restrict__ W2, const float* __restrict__ b2,
                        float* __restrict__ outp, int M) {
  int i = blockIdx.x * 128 + threadIdx.x;
  if (i >= M) return;
  const unsigned short* p = hhbf + (long)i * FFD;
  float s = b2[0];
#pragma unroll 8
  for (int j = 0; j < FFD; ++j) s += bf2f(p[j]) * W2[j];
  outp[i] = s;
}

// ---------------- host launch ----------------

extern "C" void kernel_launch(void* const* d_in, const int* in_sizes, int n_in,
                              void* d_out, int out_size, void* d_ws, size_t ws_size,
                              hipStream_t stream) {
  (void)in_sizes; (void)n_in; (void)out_size; (void)ws_size;
  const float* x        = (const float*)d_in[0];
  const float* eattr    = (const float*)d_in[1];
  const int*   eidx     = (const int*)d_in[2];
  const float* W_vert   = (const float*)d_in[3];
  const float* W_eff    = (const float*)d_in[4];
  const float* W_elin   = (const float*)d_in[5];
  const float* W_rel    = (const float*)d_in[6];
  const float* b_rel    = (const float*)d_in[7];
  const float* W_root   = (const float*)d_in[8];
  const float* b_root   = (const float*)d_in[9];
  const float* bn_g     = (const float*)d_in[10];
  const float* bn_b     = (const float*)d_in[11];
  const float* W_res    = (const float*)d_in[12];
  const float* W_h1     = (const float*)d_in[13];
  const float* b_h1     = (const float*)d_in[14];
  const float* W_h2     = (const float*)d_in[15];
  const float* b_h2     = (const float*)d_in[16];
  const int* src = eidx;
  const int* dst = eidx + NE;

  size_t off = 0;
  auto alloc = [&](size_t bytes) -> char* {
    char* p = (char*)d_ws + off;
    off += (bytes + 255) & ~(size_t)255;
    return p;
  };
  float*          h     = (float*)alloc((size_t)NN * CC * 4);
  unsigned short* hbf   = (unsigned short*)alloc((size_t)NN * CC * 2);
  unsigned short* eabf  = (unsigned short*)alloc((size_t)NE * CC * 2);
  float*          agg   = (float*)alloc((size_t)NN * CC * 4);
  unsigned short* aggbf = (unsigned short*)alloc((size_t)NN * CC * 2);
  float*          outb  = (float*)alloc((size_t)NN * CC * 4);
  unsigned short* hhbf  = (unsigned short*)alloc((size_t)NN * FFD * 2);
  unsigned short* wWe   = (unsigned short*)alloc(CC * CC * 2);   // all transposed bf16
  unsigned short* wWr   = (unsigned short*)alloc(CC * CC * 2);
  unsigned short* wWro  = (unsigned short*)alloc(CC * CC * 2);
  unsigned short* wWrs  = (unsigned short*)alloc(CC * CC * 2);
  unsigned short* wWh1  = (unsigned short*)alloc((size_t)CC * FFD * 2);
  float* bnstats = (float*)alloc(2 * CC * 4);   // sum | sumsq
  float* bscale  = (float*)alloc(CC * 4);
  float* bshift  = (float*)alloc(CC * 4);
  float* bnsum = bnstats, *bnsq = bnstats + CC;

  const int WMAT = CC * CC;                      // 16384
  const int EG = (NE / 16 + 7) / 8;              // 4688 blocks for edge GEMMs
  const int NG = (NN / 16 + 7) / 8;              // 391 blocks for node GEMMs

  k_vert<<<NN, 128, 0, stream>>>(x, W_vert, h, hbf);
  k_f2bf_t<<<(WMAT + 255) / 256, 256, 0, stream>>>(W_eff, wWe, CC);
  k_edge_ff<<<EG, 256, 0, stream>>>(eattr, wWe, eabf, NE);

  for (int l = 0; l < NLAYER; ++l) {
    k_f2bf_t<<<(WMAT + 255) / 256, 256, 0, stream>>>(W_elin + (size_t)l * WMAT, wWe, CC);
    k_f2bf_t<<<(WMAT + 255) / 256, 256, 0, stream>>>(W_rel  + (size_t)l * WMAT, wWr, CC);
    k_f2bf_t<<<(WMAT + 255) / 256, 256, 0, stream>>>(W_root + (size_t)l * WMAT, wWro, CC);
    k_f2bf_t<<<(WMAT + 255) / 256, 256, 0, stream>>>(W_res  + (size_t)l * WMAT, wWrs, CC);
    k_zero<<<(NN * CC + 255) / 256, 256, 0, stream>>>(agg, NN * CC);
    k_zero<<<1, 256, 0, stream>>>(bnstats, 2 * CC);

    k_edge_scatter<<<EG, 256, 0, stream>>>(eabf, wWe, h, src, dst, agg, NE);
    k_f2bf<<<(NN * CC + 255) / 256, 256, 0, stream>>>(agg, aggbf, NN * CC);
    k_node<<<NG, 256, 0, stream>>>(aggbf, wWr, hbf, wWro,
                                   b_rel + (size_t)l * CC, b_root + (size_t)l * CC,
                                   outb, bnsum, bnsq, NN);
    k_bn_final<<<1, 128, 0, stream>>>(bnsum, bnsq, bn_g + (size_t)l * CC,
                                      bn_b + (size_t)l * CC, bscale, bshift,
                                      1.f / (float)NN);
    k_resid<<<NG, 256, 0, stream>>>(hbf, wWrs, outb, bscale, bshift, h, hbf, NN);
  }

  k_f2bf_t<<<(CC * FFD + 255) / 256, 256, 0, stream>>>(W_h1, wWh1, FFD);
  dim3 hg(NG, FFD / 128);
  k_head<<<hg, 256, 0, stream>>>(hbf, wWh1, b_h1, hhbf, NN);
  k_head2<<<(NN + 127) / 128, 128, 0, stream>>>(hhbf, W_h2, b_h2, (float*)d_out, NN);
}
